// NodeModel_25890062860638
// MI455X (gfx1250) — matrix operand order repaired and checked by reading, not compiled, not run on hardware
//
#include <hip/hip_runtime.h>
#include <hip/hip_bf16.h>

// ---------------------------------------------------------------------------
// Types for CDNA5 WMMA (wave32): 16x16x32 bf16 -> f32
// ---------------------------------------------------------------------------
typedef __attribute__((ext_vector_type(16))) __bf16        v16bf;
typedef __attribute__((ext_vector_type(8)))  float         v8f;
typedef __attribute__((ext_vector_type(8)))  unsigned int  v8u;
typedef __attribute__((ext_vector_type(4)))  int           v4i;
typedef __attribute__((ext_vector_type(2)))  int           v2i;

#define AS1 __attribute__((address_space(1)))
#define AS3 __attribute__((address_space(3)))

// Async global->LDS path (gfx1250). Guarded per compilation pass: the host
// pass sees the fallback body (never codegen'd), the device pass uses the
// CDNA5 async-to-LDS instructions when the toolchain exposes them.
#if defined(__AMDGCN__) &&                                               \
    __has_builtin(__builtin_amdgcn_global_load_async_to_lds_b128) &&     \
    __has_builtin(__builtin_amdgcn_global_load_async_to_lds_b64)
#define HAVE_ASYNC 1
#else
#define HAVE_ASYNC 0
#endif

#if HAVE_ASYNC
#if __has_builtin(__builtin_amdgcn_s_wait_asynccnt)
#define WAIT_ASYNC(n) __builtin_amdgcn_s_wait_asynccnt(n)
#else
#define WAIT_ASYNC(n) asm volatile("s_wait_asynccnt " #n ::: "memory")
#endif
#else
#define WAIT_ASYNC(n) do {} while (0)
#endif

static __device__ __forceinline__ unsigned short f32_to_bf16(float f) {
    unsigned u = __builtin_bit_cast(unsigned, f);
    u += 0x7FFFu + ((u >> 16) & 1u);   // round-to-nearest-even
    return (unsigned short)(u >> 16);
}

// ---------------------------------------------------------------------------
// Problem constants (from reference)
// ---------------------------------------------------------------------------
#define LRES   2000
#define DFEAT  54
#define WWIN   7
#define CHN    32
#define CATW   108          // 2*DFEAT
#define CATROW 756          // WWIN*CATW
#define K1PAD  1024         // bert K
#define KFEAT  992          // 972 padded to multiple of 32
#define KY     8192         // CH*256

// ---------------------------------------------------------------------------
// Fragment helpers (ISA 7.12.2 layouts)
// ---------------------------------------------------------------------------
// A fragment: 16-bit A-matrix 16x32. Row mr; lane<16 -> K {0..7,16..23},
// lane>=16 -> K {8..15,24..31}; 2 bf16 per VGPR.
static __device__ __forceinline__ v16bf a_frag_from(const unsigned short* row32,
                                                    int hi) {
    v8u u;
#pragma unroll
    for (int v = 0; v < 8; ++v) {
        const int koff = (v < 4) ? (hi * 8 + 2 * v) : (16 + hi * 8 + 2 * (v - 4));
        u[v] = *(const unsigned*)(row32 + koff);
    }
    return __builtin_bit_cast(v16bf, u);
}
// B fragment: 32x16; lane = column; lane<16 holds K {0..7,16..23}, lane>=16 K {8..15,24..31}.
// Source is N-major (column's K values contiguous).
static __device__ __forceinline__ v16bf b_frag_from(const unsigned short* col32,
                                                    int hi) {
    v8u u;
#pragma unroll
    for (int v = 0; v < 8; ++v) u[v] = *(const unsigned*)(col32 + ((v < 4) ? (hi * 8 + 2 * v) : (16 + hi * 8 + 2 * (v - 4))));
    return __builtin_bit_cast(v16bf, u);
}

// ---------------------------------------------------------------------------
// Direct-load WMMA GEMM (used for the small-N bert GEMM):
//   C[M x Nout] = A[M x K] * B^T[N x K] + bias (+act)
//   MODE: 0 = f32 store, 1 = bf16 store, 2 = f32 + leaky_relu(0.01)
//   NTPW: n-tiles per wave (N = NTPW*64); block = 128 (4 waves)
// ---------------------------------------------------------------------------
template <int MODE, int NTPW>
__global__ __launch_bounds__(128)
void wmma_gemm_kernel(const unsigned short* __restrict__ A,
                      const unsigned short* __restrict__ BT,
                      const float* __restrict__ bias,
                      void* __restrict__ C,
                      int Nout, int K, int lda, int ldb, int ldc)
{
    const int lane = threadIdx.x & 31;
    const int wave = threadIdx.x >> 5;
    const int m0   = blockIdx.x << 4;
    const int mr   = lane & 15;
    const int hi   = lane >> 4;

    const unsigned short* arow = A + (size_t)(m0 + mr) * lda;

    v8f acc[NTPW];
#pragma unroll
    for (int i = 0; i < NTPW; ++i) acc[i] = v8f{0.f,0.f,0.f,0.f,0.f,0.f,0.f,0.f};

    for (int k0 = 0; k0 < K; k0 += 32) {
        const v16bf afrag = a_frag_from(arow + k0, hi);
        if (k0 + 32 < K) __builtin_prefetch(arow + k0 + 64, 0, 0);
#pragma unroll
        for (int i = 0; i < NTPW; ++i) {
            const int t = wave + 4 * i;
            const v16bf bfrag =
                b_frag_from(BT + (size_t)(t * 16 + mr) * ldb + k0, hi);
            acc[i] = __builtin_amdgcn_wmma_f32_16x16x32_bf16(
                false, afrag, false, bfrag, (short)0, acc[i], false, false);
        }
    }

#pragma unroll
    for (int i = 0; i < NTPW; ++i) {
        const int t = wave + 4 * i;
        const int n = t * 16 + mr;
        if (n >= Nout) continue;
        const float bv = bias[n];
#pragma unroll
        for (int r = 0; r < 8; ++r) {
            const int m = m0 + hi * 8 + r;
            float v = acc[i][r] + bv;
            if (MODE == 2) v = (v >= 0.f) ? v : 0.01f * v;
            if (MODE == 1)
                ((unsigned short*)C)[(size_t)m * ldc + n] = f32_to_bf16(v);
            else
                ((float*)C)[(size_t)m * ldc + n] = v;
        }
    }
}

// ---------------------------------------------------------------------------
// Async-to-LDS double-buffered WMMA GEMM (N fixed at 256, block = 128).
// Per K-step of 32: A tile 16x32 bf16 (1 KB), B tile 256x32 bf16 (16 KB),
// staged via GLOBAL_LOAD_ASYNC_TO_LDS (ASYNCcnt = 9 per wave per group).
// ---------------------------------------------------------------------------
static __device__ __forceinline__ void issue_group(
    const unsigned short* __restrict__ Ag,    // A + m0*lda
    const unsigned short* __restrict__ BT,
    unsigned short* sA, unsigned short* sB,
    int k0, int lda, int ldb, int tid)
{
#if HAVE_ASYNC
    // B tile: 8 x b128 (128 lanes x 16 B = 2 KB per call)
    const int bq = tid >> 2;   // row within 32-row group
    const int br = tid & 3;    // 16-B chunk within 64-B row
#pragma unroll
    for (int i = 0; i < 8; ++i) {
        const int n = i * 32 + bq;
        const unsigned short* g = BT + (size_t)n * ldb + k0 + br * 8;
        unsigned short* s = sB + n * 32 + br * 8;
        __builtin_amdgcn_global_load_async_to_lds_b128(
            (AS1 v4i*)(g), (AS3 v4i*)(s), 0, 0);
    }
    // A tile: 1 x b64 (128 lanes x 8 B = 1 KB)
    {
        const int m = tid >> 3;    // 0..15
        const int r = tid & 7;     // 8-B chunk within 64-B row
        const unsigned short* g = Ag + (size_t)m * lda + k0 + r * 4;
        unsigned short* s = sA + m * 32 + r * 4;
        __builtin_amdgcn_global_load_async_to_lds_b64(
            (AS1 v2i*)(g), (AS3 v2i*)(s), 0, 0);
    }
#else
    // Synchronous staging fallback (also what the host pass parses).
    const int bq = tid >> 2;
    const int br = tid & 3;
#pragma unroll
    for (int i = 0; i < 8; ++i) {
        const int n = i * 32 + bq;
        *(uint4*)(sB + n * 32 + br * 8) =
            *(const uint4*)(BT + (size_t)n * ldb + k0 + br * 8);
    }
    {
        const int m = tid >> 3;
        const int r = tid & 7;
        *(uint2*)(sA + m * 32 + r * 4) =
            *(const uint2*)(Ag + (size_t)m * lda + k0 + r * 4);
    }
#endif
}

template <int MODE>
__global__ __launch_bounds__(128)
void wmma_gemm_lds_kernel(const unsigned short* __restrict__ A,
                          const unsigned short* __restrict__ BT,
                          const float* __restrict__ bias,
                          void* __restrict__ C,
                          int K, int lda, int ldb, int ldc)
{
    __shared__ unsigned short sA[2][16 * 32];    //  2 x 1 KB
    __shared__ unsigned short sB[2][256 * 32];   //  2 x 16 KB

    const int tid  = threadIdx.x;
    const int lane = tid & 31;
    const int wave = tid >> 5;
    const int m0   = blockIdx.x << 4;
    const int mr   = lane & 15;
    const int hi   = lane >> 4;

    const unsigned short* Ag = A + (size_t)m0 * lda;

    v8f acc[4];
#pragma unroll
    for (int i = 0; i < 4; ++i) acc[i] = v8f{0.f,0.f,0.f,0.f,0.f,0.f,0.f,0.f};

    issue_group(Ag, BT, sA[0], sB[0], 0, lda, ldb, tid);

    int buf = 0;
    for (int k0 = 0; k0 < K; k0 += 32) {
        if (k0 + 32 < K) {
            issue_group(Ag, BT, sA[buf ^ 1], sB[buf ^ 1], k0 + 32, lda, ldb, tid);
            WAIT_ASYNC(9);    // group for current buffer has landed (in-order)
        } else {
            WAIT_ASYNC(0);
        }
        __syncthreads();

        const unsigned short* cA = sA[buf];
        const unsigned short* cB = sB[buf];
        const v16bf afrag = a_frag_from(cA + mr * 32, hi);
#pragma unroll
        for (int i = 0; i < 4; ++i) {
            const int t = wave + 4 * i;
            const v16bf bfrag = b_frag_from(cB + (t * 16 + mr) * 32, hi);
            acc[i] = __builtin_amdgcn_wmma_f32_16x16x32_bf16(
                false, afrag, false, bfrag, (short)0, acc[i], false, false);
        }
        __syncthreads();    // reads done before this buffer is refilled
        buf ^= 1;
    }

#pragma unroll
    for (int i = 0; i < 4; ++i) {
        const int t = wave + 4 * i;
        const int n = t * 16 + mr;
        const float bv = bias[n];
#pragma unroll
        for (int r = 0; r < 8; ++r) {
            const int m = m0 + hi * 8 + r;
            float v = acc[i][r] + bv;
            if (MODE == 2) v = (v >= 0.f) ? v : 0.01f * v;
            if (MODE == 1)
                ((unsigned short*)C)[(size_t)m * ldc + n] = f32_to_bf16(v);
            else
                ((float*)C)[(size_t)m * ldc + n] = v;
        }
    }
}

// ---------------------------------------------------------------------------
// fp32 -> bf16 flat convert
// ---------------------------------------------------------------------------
__global__ void cvt_bf16_kernel(const float* __restrict__ src,
                                unsigned short* __restrict__ dst, int n)
{
    const int i = blockIdx.x * blockDim.x + threadIdx.x;
    if (i < n) dst[i] = f32_to_bf16(src[i]);
}

// ---------------------------------------------------------------------------
// Transpose + convert weights into N-major bf16 with zero padding:
//   dst[n*Kpad + k] = (k<K && n<N) ? bf16(src[k*N + n]) : 0
// ---------------------------------------------------------------------------
__global__ void transpose_cvt_kernel(const float* __restrict__ src,
                                     unsigned short* __restrict__ dst,
                                     int K, int N, int Kpad, long total)
{
    const long i = (long)blockIdx.x * blockDim.x + threadIdx.x;
    if (i >= total) return;
    const int k = (int)(i % Kpad);
    const int n = (int)(i / Kpad);
    float v = 0.f;
    if (k < K && n < N) v = src[(size_t)k * N + n];
    dst[i] = f32_to_bf16(v);
}

// ---------------------------------------------------------------------------
// Window gather + 2x self-attention (7x7, 3 heads of 18) + LN + concat.
// ---------------------------------------------------------------------------
static __device__ void attn_pass(const float (&in)[WWIN][DFEAT],
                                 float (&out)[WWIN][DFEAT],
                                 float (&p)[3][WWIN][WWIN],
                                 float* mu, float* rs,
                                 const float* __restrict__ g,
                                 const float* __restrict__ b, int tid)
{
    for (int i = tid; i < 3 * 49; i += 64) {
        const int h = i / 49, r = i - h * 49;
        const int q = r / 7, k = r - q * 7;
        float s = 0.f;
#pragma unroll
        for (int d = 0; d < 18; ++d) s += in[q][h * 18 + d] * in[k][h * 18 + d];
        p[h][q][k] = s;
    }
    __syncthreads();
    for (int i = tid; i < 21; i += 64) {
        const int h = i / 7, q = i - h * 7;
        float mx = p[h][q][0];
#pragma unroll
        for (int k = 1; k < 7; ++k) mx = fmaxf(mx, p[h][q][k]);
        float sum = 0.f;
#pragma unroll
        for (int k = 0; k < 7; ++k) {
            const float e = __expf(p[h][q][k] - mx);
            p[h][q][k] = e;
            sum += e;
        }
        const float inv = 1.f / sum;
#pragma unroll
        for (int k = 0; k < 7; ++k) p[h][q][k] *= inv;
    }
    __syncthreads();
    for (int i = tid; i < WWIN * DFEAT; i += 64) {
        const int q = i / DFEAT, d = i - q * DFEAT, h = d / 18;
        float o = 0.f;
#pragma unroll
        for (int k = 0; k < 7; ++k) o += p[h][q][k] * in[k][d];
        out[q][d] = o;
    }
    __syncthreads();
    for (int q = tid; q < WWIN; q += 64) {
        float m = 0.f;
        for (int d = 0; d < DFEAT; ++d) m += out[q][d];
        m *= (1.f / DFEAT);
        float v = 0.f;
        for (int d = 0; d < DFEAT; ++d) { const float t = out[q][d] - m; v += t * t; }
        v *= (1.f / DFEAT);
        mu[q] = m;
        rs[q] = rsqrtf(v + 1e-5f);
    }
    __syncthreads();
    for (int i = tid; i < WWIN * DFEAT; i += 64) {
        const int q = i / DFEAT, d = i - q * DFEAT;
        out[q][d] = (out[q][d] - mu[q]) * rs[q] * g[d] + b[d];
    }
    __syncthreads();
}

__global__ __launch_bounds__(64)
void window_attn_kernel(const float* __restrict__ f,
                        const float* __restrict__ g1, const float* __restrict__ b1,
                        const float* __restrict__ g2, const float* __restrict__ b2,
                        float* __restrict__ cat)
{
    __shared__ float x[WWIN][DFEAT];
    __shared__ float y[WWIN][DFEAT];
    __shared__ float z[WWIN][DFEAT];
    __shared__ float p[3][WWIN][WWIN];
    __shared__ float mu[WWIN], rs[WWIN];

    const int tid = threadIdx.x;
    const int l   = blockIdx.x;
    const bool boundary = (l <= WWIN) || (l + WWIN >= LRES);

    for (int i = tid; i < WWIN * DFEAT; i += 64) {
        const int j = i / DFEAT, d = i - j * DFEAT;
        float v;
        if (boundary) v = (j == 0) ? f[l * DFEAT + d] : 0.f;
        else          v = f[(l - 4 + j) * DFEAT + d];
        x[j][d] = v;
    }
    __syncthreads();

    attn_pass(x, y, p, mu, rs, g1, b1, tid);
    attn_pass(y, z, p, mu, rs, g2, b2, tid);

    float* crow = cat + (size_t)l * CATROW;
    for (int i = tid; i < WWIN * DFEAT; i += 64) {
        const int j = i / DFEAT, d = i - j * DFEAT;
        crow[j * CATW + d]         = x[j][d];
        crow[j * CATW + DFEAT + d] = z[j][d];
    }
}

// ---------------------------------------------------------------------------
// Conv (k=3 prelu, k=5/7 relu) + maxpool + pack feats rows as bf16[992].
// ---------------------------------------------------------------------------
__global__ __launch_bounds__(128)
void conv_pack_kernel(const float* __restrict__ cat,
                      const float* __restrict__ w3, const float* __restrict__ bc3,
                      const float* __restrict__ w5, const float* __restrict__ bc5,
                      const float* __restrict__ w7, const float* __restrict__ bc7,
                      const float* __restrict__ pa,
                      unsigned short* __restrict__ feats)
{
    const int blk = blockIdx.x;
    const int l = blk >> 5;
    const int c = blk & 31;
    unsigned short* row = feats + (size_t)blk * KFEAT;

    const int w = threadIdx.x;
    if (w >= CATW) {
        const int pi = w - CATW;
        if (pi < (KFEAT - 972)) row[972 + pi] = 0;
        return;
    }

    float xin[WWIN];
    const float* xc = cat + (size_t)l * CATROW + w;
#pragma unroll
    for (int h = 0; h < WWIN; ++h) xin[h] = xc[h * CATW];

    const float a = pa[0];

    { // k = 3, PReLU
        float wk[3];
#pragma unroll
        for (int i = 0; i < 3; ++i) wk[i] = w3[c * 3 + i];
        const float bb = bc3[c];
        float v[WWIN];
#pragma unroll
        for (int h = 0; h < WWIN; ++h) {
            float s = bb;
#pragma unroll
            for (int dh = 0; dh < 3; ++dh) {
                const int hh = h + dh - 1;
                if (hh >= 0 && hh < WWIN) s += xin[hh] * wk[dh];
            }
            v[h] = (s >= 0.f) ? s : a * s;
        }
#pragma unroll
        for (int ho = 0; ho < 5; ++ho) {
            const float m = fmaxf(fmaxf(v[ho], v[ho + 1]), v[ho + 2]);
            row[ho * CATW + w] = f32_to_bf16(m);
        }
    }
    { // k = 5, ReLU
        float wk[5];
#pragma unroll
        for (int i = 0; i < 5; ++i) wk[i] = w5[c * 5 + i];
        const float bb = bc5[c];
        float v[WWIN];
#pragma unroll
        for (int h = 0; h < WWIN; ++h) {
            float s = bb;
#pragma unroll
            for (int dh = 0; dh < 5; ++dh) {
                const int hh = h + dh - 2;
                if (hh >= 0 && hh < WWIN) s += xin[hh] * wk[dh];
            }
            v[h] = fmaxf(s, 0.f);
        }
#pragma unroll
        for (int ho = 0; ho < 3; ++ho) {
            float m = v[ho];
#pragma unroll
            for (int dh = 1; dh < 5; ++dh) m = fmaxf(m, v[ho + dh]);
            row[540 + ho * CATW + w] = f32_to_bf16(m);
        }
    }
    { // k = 7, ReLU
        float wk[7];
#pragma unroll
        for (int i = 0; i < 7; ++i) wk[i] = w7[c * 7 + i];
        const float bb = bc7[c];
        float v[WWIN];
#pragma unroll
        for (int h = 0; h < WWIN; ++h) {
            float s = bb;
#pragma unroll
            for (int dh = 0; dh < 7; ++dh) {
                const int hh = h + dh - 3;
                if (hh >= 0 && hh < WWIN) s += xin[hh] * wk[dh];
            }
            v[h] = fmaxf(s, 0.f);
        }
        float m = v[0];
#pragma unroll
        for (int dh = 1; dh < 7; ++dh) m = fmaxf(m, v[dh]);
        row[864 + w] = f32_to_bf16(m);
    }
}

// ---------------------------------------------------------------------------
// Host launcher
// ---------------------------------------------------------------------------
extern "C" void kernel_launch(void* const* d_in, const int* in_sizes, int n_in,
                              void* d_out, int out_size, void* d_ws, size_t ws_size,
                              hipStream_t stream)
{
    (void)in_sizes; (void)n_in; (void)out_size; (void)ws_size;

    const float* beft    = (const float*)d_in[0];
    const float* W_bert  = (const float*)d_in[4];
    const float* b_bert  = (const float*)d_in[5];
    const float* ln_g1   = (const float*)d_in[6];
    const float* ln_b1   = (const float*)d_in[7];
    const float* ln_g2   = (const float*)d_in[8];
    const float* ln_b2   = (const float*)d_in[9];
    const float* conv1_w = (const float*)d_in[10];
    const float* conv1_b = (const float*)d_in[11];
    const float* prelu_a = (const float*)d_in[12];
    const float* conv2_w = (const float*)d_in[13];
    const float* conv2_b = (const float*)d_in[14];
    const float* conv3_w = (const float*)d_in[15];
    const float* conv3_b = (const float*)d_in[16];
    const float* W1      = (const float*)d_in[17];
    const float* b1      = (const float*)d_in[18];
    const float* W2      = (const float*)d_in[19];
    const float* b2      = (const float*)d_in[20];
    float* out = (float*)d_out;

    char* ws = (char*)d_ws;
    size_t off = 0;
    auto take = [&](size_t bytes) -> char* {
        char* p = ws + off;
        off = (off + bytes + 255) & ~(size_t)255;
        return p;
    };
    unsigned short* beft_bf = (unsigned short*)take((size_t)LRES * K1PAD * 2);
    unsigned short* wbert_t = (unsigned short*)take((size_t)64 * K1PAD * 2);
    float*          f_buf   = (float*)take((size_t)LRES * DFEAT * 4);
    float*          cat_buf = (float*)take((size_t)LRES * CATROW * 4);
    unsigned short* w1_t    = (unsigned short*)take((size_t)256 * KFEAT * 2);
    unsigned short* w2_t    = (unsigned short*)take((size_t)256 * KY * 2);
    unsigned short* y1_buf  = (unsigned short*)take((size_t)LRES * KY * 2);
    unsigned short* feats   = (unsigned short*)take((size_t)LRES * CHN * KFEAT * 2);

    // 1) precision conversions / weight transposes
    {
        const int n = LRES * K1PAD;
        cvt_bf16_kernel<<<(n + 255) / 256, 256, 0, stream>>>(beft, beft_bf, n);
    }
    {
        const long tot = 64L * K1PAD;
        transpose_cvt_kernel<<<(int)((tot + 255) / 256), 256, 0, stream>>>(
            W_bert, wbert_t, K1PAD, DFEAT, K1PAD, tot);
    }
    {
        const long tot = 256L * KFEAT;
        transpose_cvt_kernel<<<(int)((tot + 255) / 256), 256, 0, stream>>>(
            W1, w1_t, 972, 256, KFEAT, tot);
    }
    {
        const long tot = 256L * KY;
        transpose_cvt_kernel<<<(int)((tot + 255) / 256), 256, 0, stream>>>(
            W2, w2_t, KY, 256, KY, tot);
    }

    // 2) bert GEMM: f = beft @ W_bert + b   (M=2000, N=64 pad, Nout=54, K=1024)
    wmma_gemm_kernel<0, 1><<<LRES / 16, 128, 0, stream>>>(
        beft_bf, wbert_t, b_bert, f_buf, DFEAT, K1PAD, K1PAD, K1PAD, DFEAT);

    // 3) window + attention x2 + concat
    window_attn_kernel<<<LRES, 64, 0, stream>>>(f_buf, ln_g1, ln_b1, ln_g2, ln_b2, cat_buf);

    // 4) conv branches + maxpool -> feats bf16 (64000 x 992)
    conv_pack_kernel<<<LRES * CHN, 128, 0, stream>>>(
        cat_buf, conv1_w, conv1_b, conv2_w, conv2_b, conv3_w, conv3_b,
        prelu_a, feats);

    // 5) y1 = feats @ W1 + b1  (M=64000, N=256, K=992), bf16 out
    wmma_gemm_lds_kernel<1><<<(LRES * CHN) / 16, 128, 0, stream>>>(
        feats, w1_t, b1, y1_buf, KFEAT, KFEAT, KFEAT, 256);

    // 6) out = leaky_relu(y1 @ W2 + b2)  (M=2000, N=256, K=8192), f32 out
    wmma_gemm_lds_kernel<2><<<LRES / 16, 128, 0, stream>>>(
        y1_buf, w2_t, b2, out, KY, KY, KY, 256);
}
